// BiMinLSTM_901943132456
// MI455X (gfx1250) — compile-verified, matrix operand order checked by reading, not run on hardware
//
#include <hip/hip_runtime.h>
#include <stdint.h>

// Problem sizes (match reference)
#define BDIM 256
#define TDIM 512
#define EDIM 256
#define UDIM 256
#define MROWS (BDIM * TDIM)   // 131072 flattened (b,t) rows

// LDS row stride for 256 bf16 row + 8 bf16 pad = 16B pad -> banks spread,
// every row start stays 16B aligned (528 bytes).
#define LSTR 264

typedef __attribute__((ext_vector_type(16))) __bf16   v16bf;
typedef __attribute__((ext_vector_type(8)))  float    v8f;
typedef __attribute__((ext_vector_type(4)))  unsigned u32x4;
typedef __attribute__((ext_vector_type(8)))  int      i32x8;
typedef __attribute__((ext_vector_type(4)))  int      i32x4;

struct Ptr6  { const float* p[6];  };
struct Ptr12 { const float* p[12]; };

#if defined(__HIP_DEVICE_COMPILE__) && defined(__gfx1250__) && \
    __has_builtin(__builtin_amdgcn_tensor_load_to_lds) && \
    __has_builtin(__builtin_amdgcn_s_wait_tensorcnt)
#define USE_TDM 1
#else
#define USE_TDM 0
#endif

__device__ __forceinline__ float sigmoidf_(float x) {
  return 1.0f / (1.0f + __expf(-x));
}

// Load a 16-element bf16 A-fragment from a 16B-aligned (not 32B) address
// as two b128 chunks.
__device__ __forceinline__ v16bf ldA16(const __bf16* p) {
  v16bf r;
  reinterpret_cast<uint4*>(&r)[0] = *reinterpret_cast<const uint4*>(p);
  reinterpret_cast<uint4*>(&r)[1] = *reinterpret_cast<const uint4*>(p + 8);
  return r;
}

// ---------------------------------------------------------------------------
// Kernel 1: convert 12 weight matrices [K=256, N=256] fp32 row-major into
// bf16 *N-major* layout [mat][n][k] so WMMA B-fragments are 16 contiguous
// bf16 per lane. Order: mats 0..5 = U gates (f,i,c fwd; f,i,c bwd),
// mats 6..11 = W gates in the same order.
// ---------------------------------------------------------------------------
__global__ void k_pack_weights(Ptr12 src, __bf16* __restrict__ dst) {
  const int mat = blockIdx.x;       // 0..11
  const int n   = blockIdx.y;       // 0..255
  const int k   = threadIdx.x;      // 0..255
  dst[(size_t)mat * 65536 + (size_t)n * 256 + k] =
      (__bf16)src.p[mat][(size_t)k * 256 + n];
}

// ---------------------------------------------------------------------------
// Kernel 2: embedding gather + fp32 -> bf16.  X[row][e] = emb[tok[row]][e]
// ---------------------------------------------------------------------------
__global__ void k_embed(const int* __restrict__ tok,
                        const float* __restrict__ emb,
                        __bf16* __restrict__ X) {
  const size_t row = blockIdx.x;    // 0..131071 == b*T + t
  const int    e   = threadIdx.x;   // 0..255
  const int    tk  = tok[row];
  X[row * EDIM + e] = (__bf16)emb[(size_t)tk * EDIM + e];
}

// ---------------------------------------------------------------------------
// Kernel 3: input projections.  For each 16-row tile of X (bf16), compute
// P[g] = X @ W[g] + b[g] for all 6 gates with v_wmma_f32_16x16x32_bf16.
// 512 threads = 16 waves; wave w owns output column tile n0 = 16*w.
// X tile staged to LDS via TDM tensor_load_to_lds (with descriptor padding
// matching LSTR) when available, else a cooperative vector load.
// P is stored [6][T][B][U] so the scan reads contiguously per time step.
// ---------------------------------------------------------------------------
__global__ void __launch_bounds__(512)
k_proj(const __bf16* __restrict__ X,
       const __bf16* __restrict__ Wt,   // [6][N=256][K=256] bf16
       Ptr6 bias,
       float* __restrict__ P) {
  __shared__ __attribute__((aligned(32))) __bf16 sX[16 * LSTR];
  const int tid = threadIdx.x;
  const int m0  = blockIdx.x * 16;          // global row of tile

#if USE_TDM
  if (tid < 32) {   // one wave issues the TDM descriptor
    const uint64_t ga = (uint64_t)(uintptr_t)(X + (size_t)m0 * EDIM);
    u32x4 g0;
    g0[0] = 1u;                                     // count=1, user mode
    g0[1] = (unsigned)(uintptr_t)(&sX[0]);          // lds_addr
    g0[2] = (unsigned)ga;                           // global_addr[31:0]
    g0[3] = (unsigned)((ga >> 32) & 0x1FFFFFFu) | (2u << 30); // addr hi | type=2
    i32x8 g1;
    // data_size=2B (code 1), pad_enable, pad_interval=6 (128 dwords = one
    // 256-elem bf16 row), pad_amount=3 (4 dwords = 8 bf16 -> LSTR=264)
    g1[0] = (int)((1u << 16) | (1u << 20) | (6u << 22) | (3u << 25));
    g1[1] = (int)(256u << 16);                      // tensor_dim0 = 256 (lo16)
    g1[2] = (int)(((unsigned)(MROWS & 0xFFFF)) << 16); // dim0 hi=0 | dim1 lo16
    g1[3] = (int)(((unsigned)MROWS >> 16) | (256u << 16)); // dim1 hi | tile_dim0=256
    g1[4] = 16;                                     // tile_dim1 = 16 rows
    g1[5] = 256;                                    // tensor_dim0_stride = 256
    g1[6] = 0;
    g1[7] = 0;
    i32x4 z4 = {0, 0, 0, 0};
#if __clang_major__ >= 23
    i32x8 z8 = {0, 0, 0, 0, 0, 0, 0, 0};
    __builtin_amdgcn_tensor_load_to_lds(g0, g1, z4, z4, z8, 0);
#else
    __builtin_amdgcn_tensor_load_to_lds(g0, g1, z4, z4, 0);
#endif
    __builtin_amdgcn_s_wait_tensorcnt((short)0);
  }
  __syncthreads();
#else
  {
    // 512 threads copy 16 rows x 256 bf16 (8 bf16 = 16B each) with pad
    const int r  = tid >> 5;        // 0..15 row
    const int cs = tid & 31;        // 0..31 chunk of 8 elems
    *reinterpret_cast<uint4*>(&sX[r * LSTR + cs * 8]) =
        *reinterpret_cast<const uint4*>(X + ((size_t)(m0 + r)) * EDIM + cs * 8);
  }
  __syncthreads();
#endif

  const int lane = tid & 31;
  const int wv   = tid >> 5;        // 0..15
  const int n0   = wv * 16;
  const int half = lane >> 4;       // 0/1
  const int rr   = lane & 15;

  v16bf afr[8];
#pragma unroll
  for (int kc = 0; kc < 8; ++kc)
    afr[kc] = ldA16(&sX[rr * LSTR + kc * 32 + half * 16]);

  const int b_idx = m0 >> 9;        // T=512 divides tile cleanly: fixed b
  const int tbase = m0 & 511;

#pragma unroll
  for (int g = 0; g < 6; ++g) {
    const float bv = bias.p[g][n0 + rr];
    v8f acc = {bv, bv, bv, bv, bv, bv, bv, bv};
    const __bf16* bp = Wt + (size_t)g * 65536 + (size_t)(n0 + rr) * 256;
#pragma unroll
    for (int kc = 0; kc < 8; ++kc) {
      v16bf bfr = *reinterpret_cast<const v16bf*>(bp + kc * 32 + half * 16);
      acc = __builtin_amdgcn_wmma_f32_16x16x32_bf16(
          false, afr[kc], false, bfr, (short)0, acc, false, false);
    }
#pragma unroll
    for (int j = 0; j < 8; ++j) {
      const int t = tbase + j + 8 * half;
      P[(((size_t)g * TDIM + t) * BDIM + b_idx) * UDIM + n0 + rr] = acc[j];
    }
  }
}

// ---------------------------------------------------------------------------
// Kernel 4: the bidirectional MinLSTM scan.  32 persistent workgroups:
// blockIdx = dir*16 + batch_tile.  Each block owns 16 batch rows and runs
// all 512 steps; per step three [16,256]@[256,256] bf16 WMMA GEMMs
// (h @ Uf/Ui/Uc), accumulators seeded from the precomputed projections,
// then sigmoid/tanh + masked state update.  h lives in LDS (bf16, padded),
// c and h fp32 state live in registers (each wave owns a fixed 16x16 tile).
// U weights are read from L2 every step (768 KB total, L2-resident).
// ---------------------------------------------------------------------------
__global__ void __launch_bounds__(512)
k_scan(const int* __restrict__ tok,
       const __bf16* __restrict__ Ut,    // [6][N][K] bf16, gate = dir*3+g
       const float* __restrict__ P,      // [6][T][B][U] fp32
       float* __restrict__ hout) {       // [2][B][U] fp32
  __shared__ __attribute__((aligned(32))) __bf16 sH[16 * LSTR];
  __shared__ int smask[16];

  const int tid  = threadIdx.x;
  const int dir  = blockIdx.x >> 4;
  const int m0   = (blockIdx.x & 15) * 16;
  const int lane = tid & 31;
  const int wv   = tid >> 5;
  const int n0   = wv * 16;
  const int half = lane >> 4;
  const int rr   = lane & 15;
  const int myrow = 8 * half;              // row offset of this lane's C rows

  for (int i = tid; i < 16 * LSTR; i += 512) sH[i] = (__bf16)0.0f;

  float c_reg[8], h_reg[8];
#pragma unroll
  for (int j = 0; j < 8; ++j) { c_reg[j] = 0.0f; h_reg[j] = 0.0f; }

  const __bf16* U0 = Ut + (size_t)(dir * 3 + 0) * 65536 + (size_t)(n0 + rr) * 256;
  const __bf16* U1 = Ut + (size_t)(dir * 3 + 1) * 65536 + (size_t)(n0 + rr) * 256;
  const __bf16* U2 = Ut + (size_t)(dir * 3 + 2) * 65536 + (size_t)(n0 + rr) * 256;
  const float* P0 = P + (size_t)(dir * 3 + 0) * TDIM * BDIM * UDIM;
  const float* P1 = P + (size_t)(dir * 3 + 1) * TDIM * BDIM * UDIM;
  const float* P2 = P + (size_t)(dir * 3 + 2) * TDIM * BDIM * UDIM;

  for (int t = 0; t < TDIM; ++t) {
    const int tt = dir ? (TDIM - 1 - t) : t;
    if (tid < 16) smask[tid] = (tok[(size_t)(m0 + tid) * TDIM + tt] != 0);
    __syncthreads();   // h writes of prev step + masks visible

    v16bf afr[8];
#pragma unroll
    for (int kc = 0; kc < 8; ++kc)
      afr[kc] = ldA16(&sH[rr * LSTR + kc * 32 + half * 16]);

    v8f a0, a1, a2;
#pragma unroll
    for (int j = 0; j < 8; ++j) {
      const size_t off =
          ((size_t)tt * BDIM + (m0 + j + myrow)) * UDIM + n0 + rr;
      a0[j] = P0[off]; a1[j] = P1[off]; a2[j] = P2[off];
    }

#pragma unroll
    for (int kc = 0; kc < 8; ++kc) {
      v16bf b0 = *reinterpret_cast<const v16bf*>(U0 + kc * 32 + half * 16);
      a0 = __builtin_amdgcn_wmma_f32_16x16x32_bf16(
          false, afr[kc], false, b0, (short)0, a0, false, false);
      v16bf b1 = *reinterpret_cast<const v16bf*>(U1 + kc * 32 + half * 16);
      a1 = __builtin_amdgcn_wmma_f32_16x16x32_bf16(
          false, afr[kc], false, b1, (short)0, a1, false, false);
      v16bf b2 = *reinterpret_cast<const v16bf*>(U2 + kc * 32 + half * 16);
      a2 = __builtin_amdgcn_wmma_f32_16x16x32_bf16(
          false, afr[kc], false, b2, (short)0, a2, false, false);
    }

#pragma unroll
    for (int j = 0; j < 8; ++j) {
      const float f  = sigmoidf_(a0[j]);
      const float ii = sigmoidf_(a1[j]);
      const float cc = tanhf(a2[j]);
      const float cn = f * c_reg[j] + ii * cc;
      const float hn = tanhf(cn);
      if (smask[j + myrow]) { c_reg[j] = cn; h_reg[j] = hn; }
    }

    __syncthreads();   // all A-fragment reads done before overwriting h
#pragma unroll
    for (int j = 0; j < 8; ++j)
      sH[(j + myrow) * LSTR + n0 + rr] = (__bf16)h_reg[j];
  }

#pragma unroll
  for (int j = 0; j < 8; ++j)
    hout[(size_t)dir * BDIM * UDIM + (size_t)(m0 + j + myrow) * UDIM + n0 + rr] =
        h_reg[j];
}

// ---------------------------------------------------------------------------
// Kernel 5: head.  d = relu([hf|hb] @ W1 + b1); out = softmax(d @ W2 + b2).
// One block (32 threads) per batch row; tiny VALU work.
// ---------------------------------------------------------------------------
__global__ void k_head(const float* __restrict__ hout,
                       const float* __restrict__ W1, const float* __restrict__ b1,
                       const float* __restrict__ W2, const float* __restrict__ b2,
                       float* __restrict__ out) {
  __shared__ float sd[32];
  const int b = blockIdx.x;
  const int j = threadIdx.x;   // 0..31
  const float* hf = hout + (size_t)b * UDIM;
  const float* hb = hout + (size_t)BDIM * UDIM + (size_t)b * UDIM;
  float acc = b1[j];
  for (int k = 0; k < UDIM; ++k) acc += hf[k] * W1[k * 32 + j];
  for (int k = 0; k < UDIM; ++k) acc += hb[k] * W1[(UDIM + k) * 32 + j];
  sd[j] = fmaxf(acc, 0.0f);
  __syncthreads();
  if (j == 0) {
    float l[3];
#pragma unroll
    for (int c = 0; c < 3; ++c) {
      float a = b2[c];
      for (int k = 0; k < 32; ++k) a += sd[k] * W2[k * 3 + c];
      l[c] = a;
    }
    const float mx = fmaxf(l[0], fmaxf(l[1], l[2]));
    const float e0 = __expf(l[0] - mx), e1 = __expf(l[1] - mx),
                e2 = __expf(l[2] - mx);
    const float s = e0 + e1 + e2;
    out[b * 3 + 0] = e0 / s;
    out[b * 3 + 1] = e1 / s;
    out[b * 3 + 2] = e2 / s;
  }
}

// ---------------------------------------------------------------------------
// Launch.  Workspace layout (bytes):
//   X   bf16 [131072][256]                 :  67,108,864
//   Wt  bf16 [12][256][256] (U mats, W mats):  1,572,864
//   P   fp32 [6][T][B][U]                  : 805,306,368
//   H   fp32 [2][B][U]                     :     524,288
// ---------------------------------------------------------------------------
extern "C" void kernel_launch(void* const* d_in, const int* in_sizes, int n_in,
                              void* d_out, int out_size, void* d_ws, size_t ws_size,
                              hipStream_t stream) {
  (void)in_sizes; (void)n_in; (void)out_size; (void)ws_size;
  const int*   tok = (const int*)d_in[0];
  const float* emb = (const float*)d_in[1];
  const float* W1  = (const float*)d_in[20];
  const float* b1  = (const float*)d_in[21];
  const float* W2  = (const float*)d_in[22];
  const float* b2  = (const float*)d_in[23];

  auto Wg = [&](int dir, int g) { return (const float*)d_in[2 + 9 * dir + 3 * g + 0]; };
  auto Ug = [&](int dir, int g) { return (const float*)d_in[2 + 9 * dir + 3 * g + 1]; };
  auto Bg = [&](int dir, int g) { return (const float*)d_in[2 + 9 * dir + 3 * g + 2]; };

  char* ws = (char*)d_ws;
  __bf16* X  = (__bf16*)ws;
  __bf16* Wt = (__bf16*)(ws + (size_t)MROWS * EDIM * 2);
  float*  P  = (float*)(ws + (size_t)MROWS * EDIM * 2 + 12ull * 65536 * 2);
  float*  H  = (float*)((char*)P + 6ull * MROWS * UDIM * 4);

  Ptr12 pw;
  Ptr6  pb;
  for (int dir = 0; dir < 2; ++dir)
    for (int g = 0; g < 3; ++g) {
      pw.p[dir * 3 + g]     = Ug(dir, g);   // mats 0..5 : U gates
      pw.p[6 + dir * 3 + g] = Wg(dir, g);   // mats 6..11: W gates
      pb.p[dir * 3 + g]     = Bg(dir, g);
    }

  k_pack_weights<<<dim3(12, 256), 256, 0, stream>>>(pw, Wt);
  k_embed<<<MROWS, EDIM, 0, stream>>>(tok, emb, X);
  k_proj<<<MROWS / 16, 512, 0, stream>>>(X, Wt + 6ull * 65536, pb, P);
  k_scan<<<32, 512, 0, stream>>>(tok, Wt, P, H);
  k_head<<<BDIM, 32, 0, stream>>>(H, W1, b1, W2, b2, (float*)d_out);
}
